// StructurePredictor_11046655885411
// MI455X (gfx1250) — matrix operand order repaired and checked by reading, not compiled, not run on hardware
//
#include <hip/hip_runtime.h>

// ---------------------------------------------------------------------------
// StructurePredictor pipeline for MI455X (gfx1250, wave32, WMMA)
// All large GEMMs run on v_wmma_f32_16x16x32_f16 (f16 operands, f32 accum).
// ---------------------------------------------------------------------------

typedef _Float16 f16_t;
typedef __attribute__((ext_vector_type(16))) _Float16 v16h;
typedef __attribute__((ext_vector_type(8)))  float    v8f;
typedef __attribute__((ext_vector_type(4)))  int      v4i;

union Frag16 { v4i q[2]; v16h v; };

#define BB 16
#define SS 1024
#define EE 320
#define HH 8
#define DHD 40
#define MM (BB * SS)   // 16384 rows

// ---------------------------------------------------------------------------
// Pack torch-layout W [N(out) x K(in)] f32 row-major into WMMA B-fragment
// order: tiles (nt,kt) of 32(K)x16(N); per tile 32 lanes x 8 dwords.
// Lane l: n = nt*16 + (l&15); lanes 0-15 hold K=kt*32+{0..7,16..23}, lanes 16-31
// hold K=kt*32+{8..15,24..31}, two f16 per dword (low half = even K).
// ---------------------------------------------------------------------------
__global__ void pack_b_kernel(const float* __restrict__ W,
                              unsigned* __restrict__ Bp, int K, int N) {
  int d = blockIdx.x * blockDim.x + threadIdx.x;
  int total = (K * N) >> 1;
  if (d >= total) return;
  int ntk  = K >> 5;
  int t    = d >> 8;
  int rem  = d & 255;
  int lane = rem >> 3;
  int vv   = rem & 7;
  int kt   = t % ntk;
  int nt   = t / ntk;
  int n = nt * 16 + (lane & 15);
  int k = kt * 32 + ((lane >> 4) << 3) + ((vv >> 2) << 3) + (vv << 1);
  union { _Float16 h[2]; unsigned u; } pk;
  pk.h[0] = (_Float16)W[(size_t)n * K + k];
  pk.h[1] = (_Float16)W[(size_t)n * K + k + 1];
  Bp[d] = pk.u;
}

__global__ void cvt_f16_kernel(const float* __restrict__ in,
                               f16_t* __restrict__ out, int n) {
  int i = blockIdx.x * blockDim.x + threadIdx.x;
  if (i < n) out[i] = (f16_t)in[i];
}

// ---------------------------------------------------------------------------
// WMMA GEMM: out = act(A[M x K] * W^T + bias). A is f16 row-major (ldA).
// Bp is the pre-packed weight. Block = 128 threads = 4 waves; wave tile =
// 32(M) x 16*NT(N); block tile = 128 x 16*NT. grid = (M/128, N/(16*NT)).
// Writes optional f32 output (ld=N) and optional f16 output (ldH, colOff) for
// feeding the next layer / concat buffer.
// ---------------------------------------------------------------------------
template <int NT>
__global__ void __launch_bounds__(128)
gemm_wmma_kernel(const f16_t* __restrict__ A, int ldA,
                 const unsigned* __restrict__ Bp,
                 const float* __restrict__ bias,
                 int N, int K, int relu,
                 float* __restrict__ outF,
                 f16_t* __restrict__ outH, int ldH, int colOff) {
  const int lane  = threadIdx.x & 31;
  const int wave  = threadIdx.x >> 5;
  const int hi    = lane >> 4;          // half-wave select
  const int nl    = lane & 15;          // lane within 16
  const int rbase = blockIdx.x * 128 + wave * 32;
  const int cbase = blockIdx.y * (16 * NT);
  const int ntk   = K >> 5;

  v8f acc[2][NT];
#pragma unroll
  for (int mt = 0; mt < 2; ++mt)
#pragma unroll
    for (int nt = 0; nt < NT; ++nt) acc[mt][nt] = (v8f)0.0f;

  for (int kt = 0; kt < ntk; ++kt) {
    Frag16 a[2], b[NT];
#pragma unroll
    for (int mt = 0; mt < 2; ++mt) {
      // A fragment: V0-3 <- K = kt*32 + hi*8 + 0..7 ; V4-7 <- +16
      const f16_t* pa = A + (size_t)(rbase + mt * 16 + nl) * ldA + kt * 32 + hi * 8;
      a[mt].q[0] = *(const v4i*)(pa);
      a[mt].q[1] = *(const v4i*)(pa + 16);
    }
#pragma unroll
    for (int nt = 0; nt < NT; ++nt) {
      int ntG = blockIdx.y * NT + nt;
      const v4i* pb = (const v4i*)(Bp + (((size_t)ntG * ntk + kt) << 8) + lane * 8);
      b[nt].q[0] = pb[0];
      b[nt].q[1] = pb[1];
    }
#pragma unroll
    for (int mt = 0; mt < 2; ++mt)
#pragma unroll
      for (int nt = 0; nt < NT; ++nt)
        acc[mt][nt] = __builtin_amdgcn_wmma_f32_16x16x32_f16(
            false, a[mt].v, false, b[nt].v, (short)0, acc[mt][nt], false, false);
  }

  // Epilogue: C/D layout -> lane n = l&15, VGPR r -> M = r + (l>=16 ? 8 : 0)
#pragma unroll
  for (int mt = 0; mt < 2; ++mt) {
#pragma unroll
    for (int nt = 0; nt < NT; ++nt) {
      const int col = cbase + nt * 16 + nl;
      const float bv = bias[col];
#pragma unroll
      for (int r = 0; r < 8; ++r) {
        const int m = rbase + mt * 16 + hi * 8 + r;
        float val = acc[mt][nt][r] + bv;
        if (relu) val = val > 0.0f ? val : 0.0f;
        if (outF) outF[(size_t)m * N + col] = val;
        if (outH) outH[(size_t)m * ldH + colOff + col] = (f16_t)val;
      }
    }
  }
}

// ---------------------------------------------------------------------------
// Attention over the batch axis: per (s,h) the score matrix is 16x16, K=40.
// One wave per (s,h); q/k/v tiles staged in LDS as f32; softmax over m; ctx
// written in-place over q16 (disjoint footprint per block).
// ---------------------------------------------------------------------------
__global__ void __launch_bounds__(32)
attn_kernel(const f16_t* __restrict__ q, const f16_t* __restrict__ k,
            const f16_t* __restrict__ v, f16_t* __restrict__ ctx) {
  __shared__ float qs[16][41], ks[16][41], vs[16][41], sc[16][17];
  const int h = blockIdx.x & (HH - 1);
  const int s = blockIdx.x >> 3;
  const int tid = threadIdx.x;
  const size_t colBase = (size_t)h * DHD;

  for (int idx = tid; idx < 16 * DHD; idx += 32) {
    const int l = idx / DHD, d = idx % DHD;
    const size_t off = ((size_t)l * SS + s) * EE + colBase + d;
    qs[l][d] = (float)q[off];
    ks[l][d] = (float)k[off];
    vs[l][d] = (float)v[off];
  }
  __syncthreads();

  const float scale = 0.15811388300841897f;  // 1/sqrt(40)
  for (int e = tid; e < 256; e += 32) {
    const int l = e >> 4, m = e & 15;
    float sum = 0.0f;
    for (int d = 0; d < DHD; ++d) sum += qs[l][d] * ks[m][d];
    sc[l][m] = sum * scale;
  }
  __syncthreads();

  if (tid < 16) {
    float mx = -3.4e38f;
    for (int m = 0; m < 16; ++m) mx = fmaxf(mx, sc[tid][m]);
    float ex[16], sum = 0.0f;
    for (int m = 0; m < 16; ++m) { ex[m] = __expf(sc[tid][m] - mx); sum += ex[m]; }
    const float inv = 1.0f / sum;
    for (int m = 0; m < 16; ++m) sc[tid][m] = ex[m] * inv;
  }
  __syncthreads();

  for (int idx = tid; idx < 16 * DHD; idx += 32) {
    const int l = idx / DHD, d = idx % DHD;
    float sum = 0.0f;
    for (int m = 0; m < 16; ++m) sum += sc[l][m] * vs[m][d];
    ctx[((size_t)l * SS + s) * EE + colBase + d] = (f16_t)sum;
  }
}

// contact MLP final layer (N=1): prob[r] = m1[r,:] . mW2 + mb2
__global__ void prob_kernel(const f16_t* __restrict__ m1,
                            const float* __restrict__ mW2,
                            const float* __restrict__ mb2,
                            float* __restrict__ prob) {
  int r = blockIdx.x * blockDim.x + threadIdx.x;
  if (r >= MM) return;
  const f16_t* row = m1 + (size_t)r * 160;
  float acc = 0.0f;
  for (int i = 0; i < 160; ++i) acc += (float)row[i] * mW2[i];
  prob[r] = acc + mb2[0];
}

// contact_map[b,s,:] = prob[b*S+s] broadcast; float4 stores (67 MB, BW bound)
__global__ void __launch_bounds__(256)
contact_kernel(const float* __restrict__ prob, float* __restrict__ out) {
  struct alignas(16) F4 { float x, y, z, w; };
  const float p = prob[blockIdx.x];
  F4 p4; p4.x = p; p4.y = p; p4.z = p; p4.w = p;
  F4* row = (F4*)(out + (size_t)blockIdx.x * SS);
  row[threadIdx.x] = p4;
}

// position predictor final layer (N=3)
__global__ void refined_kernel(const f16_t* __restrict__ h2,
                               const float* __restrict__ pW3,
                               const float* __restrict__ pb3,
                               float* __restrict__ out) {
  int r = blockIdx.x * blockDim.x + threadIdx.x;
  if (r >= MM) return;
  const f16_t* row = h2 + (size_t)r * 160;
  float a0 = 0.0f, a1 = 0.0f, a2 = 0.0f;
  for (int i = 0; i < 160; ++i) {
    const float x = (float)row[i];
    a0 += x * pW3[i];
    a1 += x * pW3[160 + i];
    a2 += x * pW3[320 + i];
  }
  out[(size_t)r * 3 + 0] = a0 + pb3[0];
  out[(size_t)r * 3 + 1] = a1 + pb3[1];
  out[(size_t)r * 3 + 2] = a2 + pb3[2];
}

// ---------------------------------------------------------------------------
extern "C" void kernel_launch(void* const* d_in, const int* in_sizes, int n_in,
                              void* d_out, int out_size, void* d_ws, size_t ws_size,
                              hipStream_t stream) {
  (void)in_sizes; (void)n_in; (void)out_size; (void)ws_size;

  const float* x    = (const float*)d_in[0];
  const float* fpW1 = (const float*)d_in[1];  const float* fpb1 = (const float*)d_in[2];
  const float* fpW2 = (const float*)d_in[3];  const float* fpb2 = (const float*)d_in[4];
  const float* Wq   = (const float*)d_in[5];  const float* bq   = (const float*)d_in[6];
  const float* Wk   = (const float*)d_in[7];  const float* bk   = (const float*)d_in[8];
  const float* Wv   = (const float*)d_in[9];  const float* bv   = (const float*)d_in[10];
  const float* Wo   = (const float*)d_in[11]; const float* bo   = (const float*)d_in[12];
  const float* mW1  = (const float*)d_in[13]; const float* mb1  = (const float*)d_in[14];
  const float* mW2  = (const float*)d_in[15]; const float* mb2  = (const float*)d_in[16];
  const float* pW1  = (const float*)d_in[17]; const float* pb1  = (const float*)d_in[18];
  const float* pW2  = (const float*)d_in[19]; const float* pb2  = (const float*)d_in[20];
  const float* pW3  = (const float*)d_in[21]; const float* pb3  = (const float*)d_in[22];

  // Output layout: backbone | side | contact_map | refined (flat f32)
  float* outBackbone = (float*)d_out;
  float* outSide     = outBackbone + (size_t)MM * EE;
  float* outContact  = outSide + (size_t)MM * EE;
  float* outRefined  = outContact + (size_t)BB * SS * SS;

  // Workspace carve-up (reused heavily; ~75 MB)
  char* ws = (char*)d_ws;
  size_t off = 0;
  auto carve = [&](size_t bytes) {
    char* p = ws + off;
    off += (bytes + 255) & ~(size_t)255;
    return (void*)p;
  };
  f16_t* x16  = (f16_t*)carve((size_t)MM * EE * 2);        // x f16; later: attn_out
  f16_t* t1   = (f16_t*)carve((size_t)MM * EE * 2);        // fp hidden; later: h1
  f16_t* comb = (f16_t*)carve((size_t)MM * 2 * EE * 2);    // [backbone16 | side16]
  f16_t* q16  = (f16_t*)carve((size_t)MM * EE * 2);        // q; later ctx (in place)
  f16_t* k16  = (f16_t*)carve((size_t)MM * EE * 2);        // k; later m1, then h2
  f16_t* v16  = (f16_t*)carve((size_t)MM * EE * 2);        // v; later prob (f32)
  unsigned* fpW1p = (unsigned*)carve((size_t)320 * 320 * 2);
  unsigned* fpW2p = (unsigned*)carve((size_t)320 * 320 * 2);
  unsigned* Wqp   = (unsigned*)carve((size_t)320 * 320 * 2);
  unsigned* Wkp   = (unsigned*)carve((size_t)320 * 320 * 2);
  unsigned* Wvp   = (unsigned*)carve((size_t)320 * 320 * 2);
  unsigned* Wop   = (unsigned*)carve((size_t)320 * 320 * 2);
  unsigned* mW1p  = (unsigned*)carve((size_t)320 * 160 * 2);
  unsigned* pW1p  = (unsigned*)carve((size_t)640 * 320 * 2);
  unsigned* pW2p  = (unsigned*)carve((size_t)320 * 160 * 2);
  float* probBuf = (float*)v16;  // v16 free after attention

  auto pack = [&](const float* W, unsigned* Bp, int K, int N) {
    int total = (K * N) >> 1;
    pack_b_kernel<<<dim3((total + 255) / 256), dim3(256), 0, stream>>>(W, Bp, K, N);
  };
  pack(fpW1, fpW1p, 320, 320);
  pack(fpW2, fpW2p, 320, 320);
  pack(Wq, Wqp, 320, 320);
  pack(Wk, Wkp, 320, 320);
  pack(Wv, Wvp, 320, 320);
  pack(Wo, Wop, 320, 320);
  pack(mW1, mW1p, 320, 160);
  pack(pW1, pW1p, 640, 320);
  pack(pW2, pW2p, 320, 160);

  cvt_f16_kernel<<<dim3((MM * EE) / 256), dim3(256), 0, stream>>>(x, x16, MM * EE);

  const dim3 blk(128);
  const dim3 g320(MM / 128, 320 / 64);   // NT=4 -> 64-wide n-blocks
  const dim3 g160(MM / 128, 160 / 32);   // NT=2 -> 32-wide n-blocks

  // feature_processor pass 1: backbone
  gemm_wmma_kernel<4><<<g320, blk, 0, stream>>>(x16, EE, fpW1p, fpb1, 320, 320, 1,
                                                nullptr, t1, EE, 0);
  gemm_wmma_kernel<4><<<g320, blk, 0, stream>>>(t1, EE, fpW2p, fpb2, 320, 320, 0,
                                                outBackbone, comb, 2 * EE, 0);
  // feature_processor pass 2: side (reads backbone16 from comb, ldA=640)
  gemm_wmma_kernel<4><<<g320, blk, 0, stream>>>(comb, 2 * EE, fpW1p, fpb1, 320, 320, 1,
                                                nullptr, t1, EE, 0);
  gemm_wmma_kernel<4><<<g320, blk, 0, stream>>>(t1, EE, fpW2p, fpb2, 320, 320, 0,
                                                outSide, comb, 2 * EE, EE);
  // q/k/v projections from x
  gemm_wmma_kernel<4><<<g320, blk, 0, stream>>>(x16, EE, Wqp, bq, 320, 320, 0,
                                                nullptr, q16, EE, 0);
  gemm_wmma_kernel<4><<<g320, blk, 0, stream>>>(x16, EE, Wkp, bk, 320, 320, 0,
                                                nullptr, k16, EE, 0);
  gemm_wmma_kernel<4><<<g320, blk, 0, stream>>>(x16, EE, Wvp, bv, 320, 320, 0,
                                                nullptr, v16, EE, 0);
  // attention over batch axis; ctx overwrites q16 (disjoint per block)
  attn_kernel<<<dim3(SS * HH), dim3(32), 0, stream>>>(q16, k16, v16, q16);
  // output projection -> attn_out (into x16, which is now free)
  gemm_wmma_kernel<4><<<g320, blk, 0, stream>>>(q16, EE, Wop, bo, 320, 320, 0,
                                                nullptr, x16, EE, 0);
  // contact MLP
  gemm_wmma_kernel<2><<<g160, blk, 0, stream>>>(x16, EE, mW1p, mb1, 160, 320, 1,
                                                nullptr, k16, 160, 0);
  prob_kernel<<<dim3(MM / 256), dim3(256), 0, stream>>>(k16, mW2, mb2, probBuf);
  contact_kernel<<<dim3(MM), dim3(256), 0, stream>>>(probBuf, outContact);
  // position predictor on concat(backbone16, side16) = comb (K=640)
  gemm_wmma_kernel<4><<<g320, blk, 0, stream>>>(comb, 2 * EE, pW1p, pb1, 320, 640, 1,
                                                nullptr, t1, EE, 0);
  gemm_wmma_kernel<2><<<g160, blk, 0, stream>>>(t1, EE, pW2p, pb2, 160, 320, 1,
                                                nullptr, k16, 160, 0);
  refined_kernel<<<dim3(MM / 256), dim3(256), 0, stream>>>(k16, pW3, pb3, outRefined);
}
